// DTNNmsg_43078521979616
// MI455X (gfx1250) — matrix operand order repaired and checked by reading, not compiled, not run on hardware
//
#include <hip/hip_runtime.h>
#include <hip/hip_bf16.h>

// DTNN message-passing layer for MI455X (gfx1250, wave32, WMMA).
// bf16 WMMA (16x16x32, f32 accum) for all three GEMMs; weights pre-converted
// to bf16 once into d_ws; inputs converted f32->bf16 with native
// v_cvt_pk_bf16_f32 while streaming (HBM-bound target ~45us for ~1.05GB).

typedef __attribute__((ext_vector_type(16))) __bf16       v16bf;
typedef __attribute__((ext_vector_type(2)))  __bf16       v2bf;
typedef __attribute__((ext_vector_type(8)))  float        v8f;
typedef __attribute__((ext_vector_type(8)))  unsigned int v8u;

#if defined(__has_builtin)
#  if __has_builtin(__builtin_amdgcn_cvt_pk_bf16_f32)
#    define HAVE_CVT_PK_BF16 1
#  endif
#  if __has_builtin(__builtin_amdgcn_tanhf)
#    define HAVE_TANH 1
#  endif
#endif

// ---- bf16 helpers -----------------------------------------------------------
__device__ inline unsigned int pack_bf16(float lo, float hi) {
#ifdef HAVE_CVT_PK_BF16
    v2bf r = __builtin_amdgcn_cvt_pk_bf16_f32(lo, hi);
    return __builtin_bit_cast(unsigned int, r);
#else
    v2bf r;
    r[0] = (__bf16)lo;
    r[1] = (__bf16)hi;
    return __builtin_bit_cast(unsigned int, r);
#endif
}

__device__ inline float fast_tanh(float x) {
#ifdef HAVE_TANH
    return __builtin_amdgcn_tanhf(x);
#else
    return tanhf(x);
#endif
}

// ---- prep kernel: f32 weights -> bf16 in workspace -------------------------
// ws layout (u16 elements): [0..16383] Wcf(128x128) | [16384..20479] Wdf(128x32)
//                           | [20480..36863] Wfc(128x128)
__global__ void dtnn_prep_weights(const float* __restrict__ wcf,
                                  const float* __restrict__ wdf,
                                  const float* __restrict__ wfc,
                                  unsigned int* __restrict__ ws32) {
    int i = blockIdx.x * 256 + threadIdx.x;   // packs elements 2i, 2i+1
    if (i >= 18432) return;
    int e0 = 2 * i;
    float lo, hi;
    if (e0 < 16384)      { lo = wcf[e0];         hi = wcf[e0 + 1]; }
    else if (e0 < 20480) { lo = wdf[e0 - 16384]; hi = wdf[e0 - 16383]; }
    else                 { lo = wfc[e0 - 20480]; hi = wfc[e0 - 20479]; }
    ws32[i] = pack_bf16(lo, hi);
}

// ---- main kernel ------------------------------------------------------------
// Block = 256 threads = 8 waves; block handles 4 nodes (128 rows of the
// flattened [N*K, .] problem). Wave w handles rows [blk*128 + 16w, +16).
__global__ __launch_bounds__(256)
void dtnn_main(const float* __restrict__ src_h,      // [N*32, 128]
               const float* __restrict__ he,         // [N*32, 32]
               const unsigned short* __restrict__ wbf,
               const float* __restrict__ bcf,
               const float* __restrict__ bdf,
               const float* __restrict__ bfc,
               float* __restrict__ out)              // [N, 128]
{
    __shared__ __align__(16) unsigned short sWcf[128 * 128]; // [h][a] bf16
    __shared__ __align__(16) unsigned short sWdf[128 * 32];  // [h][b] bf16
    __shared__ __align__(16) unsigned short sWfc[128 * 128]; // [o][h] bf16
    __shared__ float sBcf[128], sBdf[128], sBfc[128];
    __shared__ __align__(16) unsigned short sInter[8][128 * 16]; // per-wave [h][m] bf16
    __shared__ float sPart[8][128];

    const int tid = threadIdx.x;

    // Cooperative weight load global(bf16) -> LDS as b128 copies.
    {
        const uint4* g = (const uint4*)(wbf);
        uint4* s = (uint4*)sWcf;
        #pragma unroll
        for (int i = tid; i < 2048; i += 256) s[i] = g[i];
        g = (const uint4*)(wbf + 16384);
        s = (uint4*)sWdf;
        #pragma unroll
        for (int i = tid; i < 512; i += 256) s[i] = g[i];
        g = (const uint4*)(wbf + 20480);
        s = (uint4*)sWfc;
        #pragma unroll
        for (int i = tid; i < 2048; i += 256) s[i] = g[i];
        if (tid < 128) { sBcf[tid] = bcf[tid]; sBdf[tid] = bdf[tid]; sBfc[tid] = bfc[tid]; }
    }
    __syncthreads();

    const int w  = tid >> 5;       // wave in block (0..7)
    const int l  = tid & 31;       // lane
    const int ln = l & 15;         // lane % 16
    const int hi = l >> 4;         // lane half (0/1)

    const long row = (long)blockIdx.x * 128 + w * 16 + ln; // this lane's A row

    // --- A-fragment loader: f32 global row -> bf16 WMMA A layout ------------
    // 16-bit A 16x32: lanes 0-15 hold K={kk0..kk0+7, kk0+16..kk0+23} with
    // kk0 = base + 8*hi; pairs packed (K even in [15:0]).
    auto loadAfrag = [&](const float* rowp, int kk0) -> v8u {
        const float4* q0 = (const float4*)(rowp + kk0);
        const float4* q1 = (const float4*)(rowp + kk0 + 16);
        float4 a0 = q0[0], a1 = q0[1], b0 = q1[0], b1 = q1[1];
        v8u r;
        r[0] = pack_bf16(a0.x, a0.y); r[1] = pack_bf16(a0.z, a0.w);
        r[2] = pack_bf16(a1.x, a1.y); r[3] = pack_bf16(a1.z, a1.w);
        r[4] = pack_bf16(b0.x, b0.y); r[5] = pack_bf16(b0.z, b0.w);
        r[6] = pack_bf16(b1.x, b1.y); r[7] = pack_bf16(b1.z, b1.w);
        return r;
    };

    // --- B-fragment loader from LDS weights ---------------------------------
    // 16-bit B 32x16: lane n = nbase+ln, K = {kb+8*hi..+7, kb+16+8*hi..+7}.
    auto loadBfrag = [&](const unsigned short* wlds, int ks, int nbase, int kb) -> v8u {
        const uint4* q = (const uint4*)(wlds + (nbase + ln) * ks + kb + 8 * hi);
        uint4 x = q[0], y = q[2];
        v8u r;
        r[0] = x.x; r[1] = x.y; r[2] = x.z; r[3] = x.w;
        r[4] = y.x; r[5] = y.y; r[6] = y.z; r[7] = y.w;
        return r;
    };

    // Stream this wave's 16 rows: src_h (K=128 -> 4 frags), he (K=32 -> 1 frag).
    const float* srow = src_h + row * 128;
    const float* hrow = he    + row * 32;
    v8u a1[4];
    #pragma unroll
    for (int f = 0; f < 4; ++f) a1[f] = loadAfrag(srow, 32 * f + 8 * hi);
    v8u a2 = loadAfrag(hrow, 8 * hi);

    unsigned short* inter = sInter[w];

    // GEMM1 + GEMM2 + bias + product -> bf16 intermediate in LDS [h][m].
    #pragma unroll
    for (int t = 0; t < 8; ++t) {
        v8f acc1 = {};
        #pragma unroll
        for (int f = 0; f < 4; ++f) {
            v8u b = loadBfrag(sWcf, 128, 16 * t, 32 * f);
            acc1 = __builtin_amdgcn_wmma_f32_16x16x32_bf16(
                false, __builtin_bit_cast(v16bf, a1[f]),
                false, __builtin_bit_cast(v16bf, b),
                (short)0, acc1, false, false);
        }
        v8f acc2 = {};
        {
            v8u b = loadBfrag(sWdf, 32, 16 * t, 0);
            acc2 = __builtin_amdgcn_wmma_f32_16x16x32_bf16(
                false, __builtin_bit_cast(v16bf, a2),
                false, __builtin_bit_cast(v16bf, b),
                (short)0, acc2, false, false);
        }
        float b1 = sBcf[16 * t + ln];
        float b2 = sBdf[16 * t + ln];
        // D layout: col h = 16t+ln (per lane), row m = v + 8*hi (per VGPR).
        // Store packed pairs (m, m+1 contiguous) into inter[h][m].
        unsigned int* ip = (unsigned int*)(inter + (16 * t + ln) * 16 + 8 * hi);
        #pragma unroll
        for (int v = 0; v < 8; v += 2) {
            float p0 = (acc1[v]     + b1) * (acc2[v]     + b2);
            float p1 = (acc1[v + 1] + b1) * (acc2[v + 1] + b2);
            ip[v >> 1] = pack_bf16(p0, p1);
        }
    }

    // Gather GEMM3 A-fragments (transpose via LDS): A[m=ln][k=h].
    v8u a3[4];
    #pragma unroll
    for (int f = 0; f < 4; ++f) {
        int kk0 = 32 * f + 8 * hi;
        v8u r;
        #pragma unroll
        for (int j = 0; j < 4; ++j) {
            unsigned int lo0 = inter[(kk0 + 2 * j)          * 16 + ln];
            unsigned int hi0 = inter[(kk0 + 2 * j + 1)      * 16 + ln];
            unsigned int lo1 = inter[(kk0 + 16 + 2 * j)     * 16 + ln];
            unsigned int hi1 = inter[(kk0 + 16 + 2 * j + 1) * 16 + ln];
            r[j]     = lo0 | (hi0 << 16);
            r[4 + j] = lo1 | (hi1 << 16);
        }
        a3[f] = r;
    }

    // GEMM3 + bias + tanh + per-tile row reduction.
    float part[8];
    #pragma unroll
    for (int o = 0; o < 8; ++o) {
        v8f acc = {};
        #pragma unroll
        for (int f = 0; f < 4; ++f) {
            v8u b = loadBfrag(sWfc, 128, 16 * o, 32 * f);
            acc = __builtin_amdgcn_wmma_f32_16x16x32_bf16(
                false, __builtin_bit_cast(v16bf, a3[f]),
                false, __builtin_bit_cast(v16bf, b),
                (short)0, acc, false, false);
        }
        float bo = sBfc[16 * o + ln];
        float s = 0.f;
        #pragma unroll
        for (int v = 0; v < 8; ++v) s += fast_tanh(acc[v] + bo);
        s += __shfl_xor(s, 16, 32);   // combine row halves (m 0-7 with 8-15)
        part[o] = s;
    }

    if (l < 16) {
        #pragma unroll
        for (int o = 0; o < 8; ++o) sPart[w][16 * o + l] = part[o];
    }
    __syncthreads();

    // Combine the two 16-row tiles of each node; deterministic plain stores.
    #pragma unroll
    for (int i = tid; i < 512; i += 256) {
        int nodeLocal = i >> 7;
        int c = i & 127;
        float v = sPart[2 * nodeLocal][c] + sPart[2 * nodeLocal + 1][c];
        out[((long)blockIdx.x * 4 + nodeLocal) * 128 + c] = v;
    }
}

// ---- launcher ---------------------------------------------------------------
extern "C" void kernel_launch(void* const* d_in, const int* in_sizes, int n_in,
                              void* d_out, int out_size, void* d_ws, size_t ws_size,
                              hipStream_t stream) {
    const float* src_h = (const float*)d_in[0];
    const float* he    = (const float*)d_in[1];
    const float* Wcf_w = (const float*)d_in[2];
    const float* Wcf_b = (const float*)d_in[3];
    const float* Wdf_w = (const float*)d_in[4];
    const float* Wdf_b = (const float*)d_in[5];
    const float* Wfc_w = (const float*)d_in[6];
    const float* Wfc_b = (const float*)d_in[7];
    float* out = (float*)d_out;
    unsigned int* wbf32 = (unsigned int*)d_ws;   // 36864 bf16 = 72 KB

    dtnn_prep_weights<<<(18432 + 255) / 256, 256, 0, stream>>>(Wcf_w, Wdf_w, Wfc_w, wbf32);

    // 50000 nodes * 32 neighbors = 1.6M rows; 128 rows (4 nodes) per block.
    dtnn_main<<<12500, 256, 0, stream>>>(src_h, he, (const unsigned short*)d_ws,
                                         Wcf_b, Wdf_b, Wfc_b, out);
}
